// GCN_2l_19602230739457
// MI455X (gfx1250) — matrix operand order repaired and checked by reading, not compiled, not run on hardware
//
#include <hip/hip_runtime.h>

// ---------------------------------------------------------------------------
// GCN 2-layer forward for MI455X (gfx1250, wave32, WMMA).
//
//  - Both GEMMs (x@W1 and X1@W2) run on v_wmma_f32_16x16x32_bf16:
//      one wave per 16(M) x 64(N) macro-tile (4 N-tiles), K chunked by 32.
//      A fragment loaded once per K-chunk and reused across 4 WMMAs.
//  - B operands pre-packed to exact WMMA fragment order (L2-resident).
//  - Message passing: one wave per edge; edge indices load via scalar path,
//      lanes stride the 300 features -> coalesced loads + f32 atomics into
//      an L2-resident (78.6 MB < 192 MB) accumulator.
// ---------------------------------------------------------------------------

typedef __attribute__((ext_vector_type(16))) __bf16 v16bf;
typedef __attribute__((ext_vector_type(8)))  __bf16 v8bf;
typedef __attribute__((ext_vector_type(8)))  float  v8f;

#define N_NODES   65536
#define N_EDGES   262144
#define N_GRAPHS  1024
#define F_IN      7
#define F         300     // hidden width
#define KP        320     // K padded for layer-2 A (10 chunks of 32)
#define KX        32      // K padded for layer-1 A (1 chunk of 32)
#define NT_PAD    20      // N padded 300 -> 320 = 20 tiles of 16
#define NGROUPS   5       // 4 N-tiles per wave -> 5 groups
#define KC2       10      // K chunks, layer 2
#define KC1       1       // K chunks, layer 1

#define CDIV(a, b) (((a) + (b) - 1) / (b))

// ---------------- small utility kernels ----------------

__global__ void k_fill_f32(float* __restrict__ p, float v, int n) {
  int i = blockIdx.x * blockDim.x + threadIdx.x;
  if (i < n) p[i] = v;
}

// deg[col[e]] += w[e]   (deg pre-filled with 1.0 for the self loop)
__global__ void k_deg_edges(const int* __restrict__ col,
                            const float* __restrict__ w,
                            float* __restrict__ deg) {
  int e = blockIdx.x * blockDim.x + threadIdx.x;
  if (e < N_EDGES) atomicAdd(&deg[col[e]], w[e]);
}

__global__ void k_dinv(const float* __restrict__ deg,
                       float* __restrict__ dinv,
                       float* __restrict__ selfnorm) {
  int i = blockIdx.x * blockDim.x + threadIdx.x;
  if (i < N_NODES) {
    float r = rsqrtf(deg[i]);              // deg >= 1.0 always
    dinv[i] = r;
    selfnorm[i] = r * r;                   // self-loop norm (weight 1.0)
  }
}

__global__ void k_edge_norm(const int* __restrict__ row,
                            const int* __restrict__ col,
                            const float* __restrict__ w,
                            const float* __restrict__ dinv,
                            float* __restrict__ norm) {
  int e = blockIdx.x * blockDim.x + threadIdx.x;
  if (e < N_EDGES) norm[e] = dinv[row[e]] * dinv[col[e]] * w[e];
}

// ---------------- bf16 packing ----------------

// Xbf[i, 0..31] = bf16(x[i, 0..6]), zero padded
__global__ void k_pack_x(const float* __restrict__ x, __bf16* __restrict__ Xbf) {
  int idx = blockIdx.x * blockDim.x + threadIdx.x;  // N_NODES * 32
  if (idx >= N_NODES * KX) return;
  int i = idx >> 5;
  int k = idx & 31;
  float v = (k < F_IN) ? x[i * F_IN + k] : 0.0f;
  Xbf[idx] = (__bf16)v;
}

// Abf[i, k] = bf16(relu(A1[i, 0..299] + b1)), zero-padded to KP=320
__global__ void k_relu_pack_bf16(const float* __restrict__ A1,
                                 const float* __restrict__ b1,
                                 __bf16* __restrict__ Abf) {
  int idx = blockIdx.x * blockDim.x + threadIdx.x;  // N_NODES * KP
  if (idx >= N_NODES * KP) return;
  int i = idx / KP;
  int k = idx - i * KP;
  float v = 0.0f;
  if (k < F) {
    v = A1[i * F + k] + b1[k];
    v = v > 0.0f ? v : 0.0f;
  }
  Abf[idx] = (__bf16)v;
}

// Pack a [Ksrc x 300] f32 weight into WMMA B-fragment order, bf16, padded to
// [kcn*32 x 320].  B fragment (16-bit, 32x16 KxN): lane l holds column
// n = l%16, K = kc*32 + (l>>4)*8 + j for j<8, kc*32 + 16 + (l>>4)*8 + (j-8) for j>=8.
// Layout: Bpk[((tn*kcn + kc)*32 + lane)*16 + j]
__global__ void k_pack_b(const float* __restrict__ W, int Ksrc, int kcn,
                         __bf16* __restrict__ Bpk, int total) {
  int idx = blockIdx.x * blockDim.x + threadIdx.x;
  if (idx >= total) return;
  int j    = idx & 15;
  int lane = (idx >> 4) & 31;
  int kc   = (idx >> 9) % kcn;
  int tn   = idx / (kcn * 512);
  int n = tn * 16 + (lane & 15);
  int k = kc * 32 + (j < 8 ? (lane >> 4) * 8 + j : 16 + (lane >> 4) * 8 + (j - 8));
  float v = (k < Ksrc && n < F) ? W[k * F + n] : 0.0f;
  Bpk[idx] = (__bf16)v;
}

// ---------------- WMMA GEMM: [M x kcn*32]bf16 @ packed B -> f32 [M x 300] ---
// One wave per block; each wave computes a 16(M) x 64(N) macro-tile.

template <int KCN>
__global__ __launch_bounds__(32)
void k_wmma_gemm(const __bf16* __restrict__ Abf, int astride,
                 const __bf16* __restrict__ Bpk,
                 float* __restrict__ C) {
  const int lane = threadIdx.x;
  const int tm = blockIdx.x / NGROUPS;        // M tile (16 rows)
  const int tg = blockIdx.x - tm * NGROUPS;   // group of 4 N-tiles
  const int half = lane >> 4;                 // 0: lanes 0-15, 1: lanes 16-31
  const int m = tm * 16 + (lane & 15);
  const __bf16* arow = Abf + (size_t)m * astride;
  const __bf16* bbase = Bpk + (size_t)(tg * 4) * KCN * 512 + lane * 16;

  v8f acc[4] = {};
#pragma unroll
  for (int kc = 0; kc < KCN; ++kc) {
    // A fragment (16-bit A 16x32 layout): half 0 holds K {0..7,16..23},
    // half 1 holds K {8..15,24..31} of this chunk -> two 16B loads.
    v8bf alo = *(const v8bf*)(arow + kc * 32 + half * 8);
    v8bf ahi = *(const v8bf*)(arow + kc * 32 + 16 + half * 8);
    v16bf a;
#pragma unroll
    for (int j = 0; j < 8; ++j) { a[j] = alo[j]; a[8 + j] = ahi[j]; }

#pragma unroll
    for (int t = 0; t < 4; ++t) {
      // B fragment: pre-packed contiguous 32B per lane; tile stride KCN*512.
      v16bf b = *(const v16bf*)(bbase + (size_t)t * KCN * 512 + kc * 512);
      acc[t] = __builtin_amdgcn_wmma_f32_16x16x32_bf16(
          false, a, false, b, (short)0, acc[t], false, false);
    }
  }

  // D layout (32-bit C/D 16x16): VGPR r -> row r + 8*half, col = lane%16.
#pragma unroll
  for (int t = 0; t < 4; ++t) {
    const int colN = (tg * 4 + t) * 16 + (lane & 15);
    if (colN < F) {
#pragma unroll
      for (int r = 0; r < 8; ++r) {
        int rowM = tm * 16 + half * 8 + r;
        C[(size_t)rowM * F + colN] = acc[t][r];
      }
    }
  }
}

// ---------------- message passing ----------------

// A[i,f] = selfnorm[i] * H[i,f]   (initializes the accumulator)
__global__ void k_self_init(const float* __restrict__ selfnorm,
                            const float* __restrict__ H,
                            float* __restrict__ A) {
  int idx = blockIdx.x * blockDim.x + threadIdx.x;
  if (idx >= N_NODES * F) return;
  int i = idx / F;
  A[idx] = selfnorm[i] * H[idx];
}

// One wave per edge: indices hit the scalar path (blockIdx is uniform),
// lanes stride the 300 features -> coalesced loads + atomics.
__global__ __launch_bounds__(32)
void k_edge_agg(const int* __restrict__ row,
                const int* __restrict__ col,
                const float* __restrict__ norm,
                const float* __restrict__ H,
                float* __restrict__ A) {
  const int e = blockIdx.x;
  const int r = row[e];
  const int c = col[e];
  const float w = norm[e];
  const float* hs = H + (size_t)r * F;
  float* ad = A + (size_t)c * F;
  for (int f = threadIdx.x; f < F; f += 32)
    atomicAdd(&ad[f], w * hs[f]);
}

// ---------------- pooling + head ----------------

// One wave per node; batch id via scalar path; relu+bias fused.
__global__ __launch_bounds__(32)
void k_pool(const float* __restrict__ A2,
            const float* __restrict__ b2,
            const int* __restrict__ batch,
            float* __restrict__ pooled) {
  const int i = blockIdx.x;
  const int g = batch[i];
  const float* src = A2 + (size_t)i * F;
  float* dst = pooled + (size_t)g * F;
  for (int f = threadIdx.x; f < F; f += 32) {
    float v = src[f] + b2[f];
    v = v > 0.0f ? v : 0.0f;
    atomicAdd(&dst[f], v);
  }
}

__global__ void k_counts(const int* __restrict__ batch,
                         float* __restrict__ counts) {
  int i = blockIdx.x * blockDim.x + threadIdx.x;
  if (i < N_NODES) atomicAdd(&counts[batch[i]], 1.0f);
}

__global__ void k_final(const float* __restrict__ pooled,
                        const float* __restrict__ counts,
                        const float* __restrict__ Wf,
                        const float* __restrict__ bfv,
                        float* __restrict__ out) {
  int idx = blockIdx.x * blockDim.x + threadIdx.x;  // 1024*2
  if (idx >= N_GRAPHS * 2) return;
  int g = idx >> 1;
  int c = idx & 1;
  float inv = 1.0f / fmaxf(counts[g], 1.0f);
  float s = bfv[c];
  for (int f = 0; f < F; ++f)
    s = fmaf(pooled[g * F + f] * inv, Wf[f * 2 + c], s);
  out[idx] = s;
}

// ---------------- launcher ----------------

extern "C" void kernel_launch(void* const* d_in, const int* in_sizes, int n_in,
                              void* d_out, int out_size, void* d_ws, size_t ws_size,
                              hipStream_t stream) {
  const float* x    = (const float*)d_in[0];
  const int*   ei   = (const int*)  d_in[1];   // [2, E]
  const float* ew   = (const float*)d_in[2];
  const int*   bat  = (const int*)  d_in[3];
  const float* W1   = (const float*)d_in[4];
  const float* b1   = (const float*)d_in[5];
  const float* W2   = (const float*)d_in[6];
  const float* b2   = (const float*)d_in[7];
  const float* Wf   = (const float*)d_in[8];
  const float* bfv  = (const float*)d_in[9];
  float* out = (float*)d_out;

  const int* row = ei;
  const int* col = ei + N_EDGES;

  // workspace carve-up (~207 MB); every buffer fully written before read.
  char* p = (char*)d_ws;
  float* buf0     = (float*)p;  p += (size_t)N_NODES * F * 4;      // 78.6 MB
  float* buf1     = (float*)p;  p += (size_t)N_NODES * F * 4;      // 78.6 MB
  __bf16* Abf     = (__bf16*)p; p += (size_t)N_NODES * KP * 2;     // 41.9 MB
  __bf16* Xbf     = (__bf16*)p; p += (size_t)N_NODES * KX * 2;     // 4.2 MB
  __bf16* B1pk    = (__bf16*)p; p += (size_t)NT_PAD * KC1 * 512 * 2;
  __bf16* B2pk    = (__bf16*)p; p += (size_t)NT_PAD * KC2 * 512 * 2;
  float* deg      = (float*)p;  p += (size_t)N_NODES * 4;
  float* dinv     = (float*)p;  p += (size_t)N_NODES * 4;
  float* selfnorm = (float*)p;  p += (size_t)N_NODES * 4;
  float* norm     = (float*)p;  p += (size_t)N_EDGES * 4;
  float* pooled   = (float*)p;  p += (size_t)N_GRAPHS * F * 4;
  float* counts   = (float*)p;  p += (size_t)N_GRAPHS * 4;
  (void)ws_size; (void)in_sizes; (void)n_in; (void)out_size;

  const int B = 256;
  const int wmma_grid = (N_NODES / 16) * NGROUPS;   // 20480 waves

  // 1. normalization precompute
  k_fill_f32 <<<CDIV(N_NODES, B), B, 0, stream>>>(deg, 1.0f, N_NODES);
  k_deg_edges<<<CDIV(N_EDGES, B), B, 0, stream>>>(col, ew, deg);
  k_dinv     <<<CDIV(N_NODES, B), B, 0, stream>>>(deg, dinv, selfnorm);
  k_edge_norm<<<CDIV(N_EDGES, B), B, 0, stream>>>(row, col, ew, dinv, norm);

  // 2. pack operands for layer-1 WMMA
  k_pack_x<<<CDIV(N_NODES * KX, B), B, 0, stream>>>(x, Xbf);
  k_pack_b<<<CDIV(NT_PAD * KC1 * 512, B), B, 0, stream>>>(
      W1, F_IN, KC1, B1pk, NT_PAD * KC1 * 512);

  // 3. layer 1: H1 = x@W1 (WMMA); A1 = propagate(H1)
  k_wmma_gemm<KC1><<<wmma_grid, 32, 0, stream>>>(Xbf, KX, B1pk, buf0);
  k_self_init<<<CDIV(N_NODES * F, B), B, 0, stream>>>(selfnorm, buf0, buf1);
  k_edge_agg <<<N_EDGES, 32, 0, stream>>>(row, col, norm, buf0, buf1);

  // 4. relu+bias -> bf16 padded; pack W2 to fragment order
  k_relu_pack_bf16<<<CDIV(N_NODES * KP, B), B, 0, stream>>>(buf1, b1, Abf);
  k_pack_b<<<CDIV(NT_PAD * KC2 * 512, B), B, 0, stream>>>(
      W2, F, KC2, B2pk, NT_PAD * KC2 * 512);

  // 5. layer 2: H2 = X1@W2 (WMMA, 40 wmma/wave); A2 = propagate(H2)
  k_wmma_gemm<KC2><<<wmma_grid, 32, 0, stream>>>(Abf, KP, B2pk, buf0);
  k_self_init<<<CDIV(N_NODES * F, B), B, 0, stream>>>(selfnorm, buf0, buf1);
  k_edge_agg <<<N_EDGES, 32, 0, stream>>>(row, col, norm, buf0, buf1);

  // 6. mean pool (relu + bias fused)
  k_fill_f32<<<CDIV(N_GRAPHS * F, B), B, 0, stream>>>(pooled, 0.0f, N_GRAPHS * F);
  k_fill_f32<<<CDIV(N_GRAPHS, B), B, 0, stream>>>(counts, 0.0f, N_GRAPHS);
  k_pool    <<<N_NODES, 32, 0, stream>>>(buf1, b2, bat, pooled);
  k_counts  <<<CDIV(N_NODES, B), B, 0, stream>>>(bat, counts);

  // 7. head
  k_final<<<CDIV(N_GRAPHS * 2, B), B, 0, stream>>>(pooled, counts, Wf, bfv, out);
}
